// AdaptiveMixing_44169443672621
// MI455X (gfx1250) — matrix operand order repaired and checked by reading, not compiled, not run on hardware
//
#include <hip/hip_runtime.h>
#include <math.h>

typedef _Float16 h16;
typedef __attribute__((ext_vector_type(16))) _Float16 v16h;
typedef __attribute__((ext_vector_type(8)))  _Float16 v8h;
typedef __attribute__((ext_vector_type(4)))  _Float16 v4h;
typedef __attribute__((ext_vector_type(8)))  float    v8f;
typedef __attribute__((ext_vector_type(4)))  float    v4f;

#define IN_DIM   256
#define IN_PTS   64
#define OUT_DIM  256
#define OUT_PTS  64
#define QDIM     256
#define PGI      64
#define CM_SZ    (IN_DIM * OUT_DIM)     /* 65536 */
#define SM_SZ    (IN_PTS * OUT_PTS)     /* 4096  */
#define TOT      (CM_SZ + SM_SZ)        /* 69632 */
#define KTOT     (OUT_PTS * OUT_DIM)    /* 16384 */
#define CHUNK    512                    /* tokens per pipeline chunk */

__device__ __forceinline__ float gelu_exact(float z) {
  return 0.5f * z * (1.0f + erff(z * 0.7071067811865475f));
}

union F16x16 { v16h v; v8h h[2]; };

// A fragment: halves 0..7 at row[0..7], halves 8..15 at row[16..23]
// caller passes row_ptr + 8*(lane>>4)
__device__ __forceinline__ v16h ld_frag_a(const h16* p) {
  F16x16 u; u.h[0] = *(const v8h*)p; u.h[1] = *(const v8h*)(p + 16); return u.v;
}
// B fragment: halves 0..7 at row[0..7], halves 8..15 at row[16..23]; caller passes row_ptr + 8*(lane>>4)
__device__ __forceinline__ v16h ld_frag_b(const h16* p) {
  F16x16 u; u.h[0] = *(const v8h*)p; u.h[1] = *(const v8h*)(p + 16); return u.v;
}

// CDNA5 async global->LDS copy, 16 bytes per lane (tracked by ASYNCcnt)
__device__ __forceinline__ void async_copy16(void* lds, const void* g) {
  unsigned l = (unsigned)(unsigned long long)lds;  // LDS byte offset = addr[31:0]
  asm volatile("global_load_async_to_lds_b128 %0, %1, off" : : "v"(l), "v"(g) : "memory");
}
#define WAIT_ASYNC(n) asm volatile("s_wait_asynccnt %0" : : "n"(n) : "memory")

// ======================= Kcvt: weight precasts =======================
__global__ __launch_bounds__(256) void k_cvt_w2(const float* __restrict__ w2,
                                                h16* __restrict__ w2h) {
  const int i = (blockIdx.x * 256 + threadIdx.x) * 4;   // TOT*PGI divisible by 1024
  v4f v = *(const v4f*)(w2 + i);
  v4h h = { (h16)v.x, (h16)v.y, (h16)v.z, (h16)v.w };
  *(v4h*)(w2h + i) = h;
}

// wo[r][q*256+o] -> woh[r][o*64+q]  (so K3's B-fragments are K-contiguous)
__global__ __launch_bounds__(256) void k_cvt_wo(const float* __restrict__ wo,
                                                h16* __restrict__ woh) {
  const int e = (blockIdx.x * 256 + threadIdx.x) * 4;   // 4 consecutive q
  const int r = e >> 14;
  const int j = e & 16383;
  const int o = j >> 6;
  const int q = j & 63;
  const float* src = wo + r * 16384 + q * 256 + o;      // stride 256 along q
  v4h h = { (h16)src[0], (h16)src[256], (h16)src[512], (h16)src[768] };
  *(v4h*)(woh + e) = h;
}

// ======================= K0: H = f16(LN(query) @ w1^T + b1) =======================
__global__ __launch_bounds__(256) void k_gen_h(const float* __restrict__ query,
                                               const float* __restrict__ ln_g,
                                               const float* __restrict__ ln_b,
                                               const float* __restrict__ w1,
                                               const float* __restrict__ b1,
                                               h16* __restrict__ Hh) {
  __shared__ float qn[QDIM];
  __shared__ float red[16];
  const int tid = threadIdx.x, lane = tid & 31, wave = tid >> 5;
  const int token = blockIdx.x;

  float qv = query[(size_t)token * QDIM + tid];
  float s = qv, s2 = qv * qv;
  #pragma unroll
  for (int off = 16; off > 0; off >>= 1) {
    s  += __shfl_xor(s,  off, 32);
    s2 += __shfl_xor(s2, off, 32);
  }
  if (lane == 0) { red[wave] = s; red[8 + wave] = s2; }
  __syncthreads();
  float ts = 0.f, ts2 = 0.f;
  #pragma unroll
  for (int w = 0; w < 8; ++w) { ts += red[w]; ts2 += red[8 + w]; }
  const float mu   = ts  * (1.f / QDIM);
  const float var  = ts2 * (1.f / QDIM) - mu * mu;
  const float rstd = rsqrtf(var + 1e-5f);
  qn[tid] = (qv - mu) * rstd * ln_g[tid] + ln_b[tid];
  __syncthreads();

  if (tid < PGI) {
    float acc = b1[tid];
    const float* wr = w1 + tid * QDIM;
    #pragma unroll 4
    for (int c = 0; c < QDIM; ++c) acc += qn[c] * wr[c];
    Hh[(size_t)token * PGI + tid] = (h16)acc;
  }
}

// ======================= K1: P = f16(H @ w2^T + b2), WMMA GEMM =======================
// blocks 0..511   : cm part -> cmT[t][o][c]   (t-tile = blk>>4, o-group16 = blk&15)
// blocks 512..543 : sm part -> smP[t][q*64+p] (t-tile = blk-512)
__global__ __launch_bounds__(256) void k_gen_params(const h16* __restrict__ w2h,
                                                    const float* __restrict__ b2,
                                                    const h16* __restrict__ Hh,
                                                    h16* __restrict__ cmT,
                                                    h16* __restrict__ smP,
                                                    int tok0) {
  __shared__ h16 STG[16 * 16 * 64];   // [t16][o16][c64] staging, 32 KB
  const int tid = threadIdx.x, lane = tid & 31, wave = tid >> 5;
  const int nloc = lane & 15, hi = lane >> 4;
  const int blk = blockIdx.x;
  const int tt = (blk < 512) ? (blk >> 4) : (blk - 512);

  // B = H^T fragments (lane n = token, halves = table-order k) — loaded once
  const h16* hrow = Hh + (size_t)(tok0 + tt * 16 + nloc) * PGI;
  const v16h bk0 = ld_frag_b(hrow + hi * 8);
  const v16h bk1 = ld_frag_b(hrow + 32 + hi * 8);

  if (blk < 512) {
    const int og = blk & 15;            // group of 16 o-columns
    #pragma unroll 1
    for (int cb = 0; cb < 4; ++cb) {    // 64 c's per staging pass
      #pragma unroll 1
      for (int j = 0; j < 8; ++j) {
        const int c = cb * 64 + wave * 8 + j;
        const int idxbase = c * OUT_DIM + og * 16;      // 16 contiguous idx = 16 o's
        const h16* arow = w2h + (size_t)(idxbase + nloc) * PGI;
        v8f acc = {};
        acc = __builtin_amdgcn_wmma_f32_16x16x32_f16(false, ld_frag_a(arow + hi * 8),
                                                     false, bk0, (short)0, acc, false, false);
        acc = __builtin_amdgcn_wmma_f32_16x16x32_f16(false, ld_frag_a(arow + 32 + hi * 8),
                                                     false, bk1, (short)0, acc, false, false);
        v4f b2lo = *(const v4f*)(b2 + idxbase + hi * 8);
        v4f b2hi = *(const v4f*)(b2 + idxbase + hi * 8 + 4);
        const float badd[8] = { b2lo.x, b2lo.y, b2lo.z, b2lo.w,
                                b2hi.x, b2hi.y, b2hi.z, b2hi.w };
        // D tile: m = o-local (v + 8*hi), n = t-local (nloc); scatter into STG
        #pragma unroll
        for (int v = 0; v < 8; ++v)
          STG[(nloc * 16 + hi * 8 + v) * 64 + wave * 8 + j] = (h16)(acc[v] + badd[v]);
      }
      __syncthreads();
      // cooperative, cacheline-sized stores of cmT rows [t][o][c-block]
      {
        const int t_l = tid >> 4, o_l = tid & 15;
        h16* dst = cmT + (size_t)(tt * 16 + t_l) * CM_SZ
                       + (size_t)(og * 16 + o_l) * IN_DIM + cb * 64;
        const h16* srcp = STG + tid * 64;
        #pragma unroll
        for (int s = 0; s < 8; ++s)
          *(v8h*)(dst + s * 8) = *(const v8h*)(srcp + s * 8);
      }
      __syncthreads();
    }
  } else {
    // sm part: 4096 idx = 256 tiles of 16; 32 tiles per wave; natural layout out
    #pragma unroll 1
    for (int j = 0; j < 32; ++j) {
      const int tile = wave * 32 + j;
      const int li0  = tile * 16;
      const int idxbase = CM_SZ + li0;
      const h16* arow = w2h + (size_t)(idxbase + nloc) * PGI;
      v8f acc = {};
      acc = __builtin_amdgcn_wmma_f32_16x16x32_f16(false, ld_frag_a(arow + hi * 8),
                                                   false, bk0, (short)0, acc, false, false);
      acc = __builtin_amdgcn_wmma_f32_16x16x32_f16(false, ld_frag_a(arow + 32 + hi * 8),
                                                   false, bk1, (short)0, acc, false, false);
      v4f b2lo = *(const v4f*)(b2 + idxbase + hi * 8);
      v4f b2hi = *(const v4f*)(b2 + idxbase + hi * 8 + 4);
      const float badd[8] = { b2lo.x, b2lo.y, b2lo.z, b2lo.w,
                              b2hi.x, b2hi.y, b2hi.z, b2hi.w };
      union { v8h v; h16 e[8]; } pk;
      #pragma unroll
      for (int v = 0; v < 8; ++v) pk.e[v] = (h16)(acc[v] + badd[v]);
      *(v8h*)(smP + (size_t)(tt * 16 + nloc) * SM_SZ + li0 + hi * 8) = pk.v;
    }
  }
}

// ======================= K2: fused channel + spatial mixing =======================
__global__ __launch_bounds__(256) void k_mix(const float* __restrict__ sampled,
                                             const h16* __restrict__ cmT,
                                             const h16* __restrict__ smP,
                                             const float* __restrict__ m_beta,
                                             const float* __restrict__ s_beta,
                                             h16* __restrict__ o2t,
                                             int tok0) {
  __shared__ h16 XH[IN_PTS * IN_DIM];       // 32 KB : x tile f16 [p][c]
  __shared__ h16 SLAB[2][32 * IN_DIM];      // 32 KB : double-buffered cm slab [o][c]
  __shared__ h16 SMX[OUT_PTS * IN_PTS];     //  8 KB : sm [q][p]
  __shared__ h16 O1B[32 * IN_PTS];          //  4 KB : out1 block [o][p]

  const int tid = threadIdx.x, lane = tid & 31, wave = tid >> 5;
  const int nloc = lane & 15, hi = lane >> 4;
  const int tl = blockIdx.x;                // chunk-local token
  const int tok = tok0 + tl;

  // async: sm (8 KB) and slab 0 (16 KB), contiguous in the cmT layout
  {
    const h16* g = smP + (size_t)tl * SM_SZ + tid * 16;
    async_copy16(SMX + tid * 16, g);
    async_copy16(SMX + tid * 16 + 8, g + 8);
  }
  {
    const h16* g = cmT + (size_t)tl * CM_SZ + tid * 32;
    h16* l = SLAB[0] + tid * 32;
    #pragma unroll
    for (int s = 0; s < 4; ++s) async_copy16(l + s * 8, g + s * 8);
  }
  // x tile: load f32, convert to f16 into LDS
  {
    const float* xg = sampled + (size_t)tok * (IN_PTS * IN_DIM) + tid * 64;
    h16* xl = XH + tid * 64;
    #pragma unroll
    for (int i = 0; i < 16; ++i) {
      v4f v = *(const v4f*)(xg + i * 4);
      v4h h = { (h16)v.x, (h16)v.y, (h16)v.z, (h16)v.w };
      *(v4h*)(xl + i * 4) = h;
    }
  }

  const int rt = wave & 3;    // p-tile (channel) / q-tile (spatial)
  const int ct = wave >> 2;   // o-half of the 32-wide block

  #pragma unroll 1
  for (int ob = 0; ob < 8; ++ob) {
    const int o0 = ob * 32;
    const h16* cur = SLAB[ob & 1];
    if (ob < 7) {   // prefetch next slab while computing this one
      const h16* g = cmT + (size_t)tl * CM_SZ + (size_t)(o0 + 32) * IN_DIM + tid * 32;
      h16* l = SLAB[(ob + 1) & 1] + tid * 32;
      #pragma unroll
      for (int s = 0; s < 4; ++s) async_copy16(l + s * 8, g + s * 8);
      WAIT_ASYNC(4);          // everything except the 4 just-issued is complete
    } else {
      WAIT_ASYNC(0);
    }
    __syncthreads();

    // channel mixing: out1 blk[64 p][32 o] = GELU(x @ cm + m_beta)
    {
      v8f acc = {};
      #pragma unroll
      for (int kb = 0; kb < 8; ++kb) {
        v16h a = ld_frag_a(XH + (rt * 16 + nloc) * IN_DIM + kb * 32 + hi * 8);
        v16h b = ld_frag_b(cur + (ct * 16 + nloc) * IN_DIM + kb * 32 + hi * 8);
        acc = __builtin_amdgcn_wmma_f32_16x16x32_f16(false, a, false, b,
                                                     (short)0, acc, false, false);
      }
      const float mb = m_beta[o0 + ct * 16 + nloc];
      union { v8h v; h16 e[8]; } pk;
      #pragma unroll
      for (int v = 0; v < 8; ++v) pk.e[v] = (h16)gelu_exact(acc[v] + mb);
      *(v8h*)(O1B + (ct * 16 + nloc) * IN_PTS + rt * 16 + hi * 8) = pk.v;
    }
    __syncthreads();

    // spatial mixing: out2 blk[64 q][32 o] = GELU(sm @ out1 + s_beta) -> o2t[t][o*64+q]
    {
      v8f acc = {};
      #pragma unroll
      for (int kb = 0; kb < 2; ++kb) {
        v16h a = ld_frag_a(SMX + (rt * 16 + nloc) * IN_PTS + kb * 32 + hi * 8);
        v16h b = ld_frag_b(O1B + (ct * 16 + nloc) * IN_PTS + kb * 32 + hi * 8);
        acc = __builtin_amdgcn_wmma_f32_16x16x32_f16(false, a, false, b,
                                                     (short)0, acc, false, false);
      }
      const int o = o0 + ct * 16 + nloc;
      union { v8h v; h16 e[8]; } pk;
      #pragma unroll
      for (int v = 0; v < 8; ++v) {
        const int q = rt * 16 + hi * 8 + v;
        pk.e[v] = (h16)gelu_exact(acc[v] + s_beta[q]);
      }
      *(v8h*)(o2t + (size_t)tl * KTOT + (size_t)o * 64 + rt * 16 + hi * 8) = pk.v;
    }
  }
}

// ======================= K3: Y = OUT2 @ wo_perm^T + bo, WMMA GEMM =======================
// grid: (t-tile 32) x (r-tile 16); 8-way K-split across waves + LDS reduction
__global__ __launch_bounds__(256) void k_proj(const h16* __restrict__ o2t,
                                              const h16* __restrict__ woh,
                                              const float* __restrict__ bo,
                                              float* __restrict__ out,
                                              int tok0) {
  __shared__ float RED[8 * 256];
  const int tid = threadIdx.x, lane = tid & 31, wave = tid >> 5;
  const int nloc = lane & 15, hi = lane >> 4;
  const int tt = blockIdx.x >> 4;
  const int r0 = (blockIdx.x & 15) * 16;

  const h16* arow = o2t + (size_t)(tt * 16 + nloc) * KTOT;
  const h16* brow = woh + (size_t)(r0 + nloc) * KTOT;

  v8f acc = {};
  const int kb0 = wave * 64;
  #pragma unroll 1
  for (int kb = kb0; kb < kb0 + 64; ++kb) {
    const int k0 = kb * 32;
    __builtin_prefetch(arow + k0 + 512, 0, 0);
    v16h a = ld_frag_a(arow + k0 + hi * 8);
    v16h b = ld_frag_b(brow + k0 + hi * 8);
    acc = __builtin_amdgcn_wmma_f32_16x16x32_f16(false, a, false, b,
                                                 (short)0, acc, false, false);
  }
  #pragma unroll
  for (int v = 0; v < 8; ++v) RED[wave * 256 + (hi * 8 + v) * 16 + nloc] = acc[v];
  __syncthreads();

  const int m = tid >> 4, n = tid & 15;
  float sum = bo[r0 + n];
  #pragma unroll
  for (int w = 0; w < 8; ++w) sum += RED[w * 256 + m * 16 + n];
  out[(size_t)(tok0 + tt * 16 + m) * OUT_DIM + r0 + n] = sum;
}

// ======================= host side =======================
extern "C" void kernel_launch(void* const* d_in, const int* in_sizes, int n_in,
                              void* d_out, int out_size, void* d_ws, size_t ws_size,
                              hipStream_t stream) {
  const float* sampled = (const float*)d_in[0];
  const float* query   = (const float*)d_in[1];
  const float* ln_g    = (const float*)d_in[2];
  const float* ln_b    = (const float*)d_in[3];
  const float* w1      = (const float*)d_in[4];
  const float* b1      = (const float*)d_in[5];
  const float* w2      = (const float*)d_in[6];
  const float* b2      = (const float*)d_in[7];
  const float* m_beta  = (const float*)d_in[8];
  const float* s_beta  = (const float*)d_in[9];
  const float* wo      = (const float*)d_in[10];
  const float* bo      = (const float*)d_in[11];
  float* out = (float*)d_out;

  const int BT  = in_sizes[0] / (IN_PTS * IN_DIM);
  const int nch = BT / CHUNK;

  // workspace carve-up (all sizes multiples of 256 B); ~102 MB total
  char* p = (char*)d_ws;
  h16* Hh   = (h16*)p;  p += (size_t)BT * PGI * 2;
  h16* w2h  = (h16*)p;  p += (size_t)TOT * PGI * 2;
  h16* woh  = (h16*)p;  p += (size_t)OUT_DIM * KTOT * 2;
  h16* cmT  = (h16*)p;  p += (size_t)CHUNK * CM_SZ * 2;
  h16* smP  = (h16*)p;  p += (size_t)CHUNK * SM_SZ * 2;
  h16* o2t  = (h16*)p;  p += (size_t)CHUNK * KTOT * 2;

  hipLaunchKernelGGL(k_cvt_w2, dim3((TOT * PGI) / 1024), dim3(256), 0, stream, w2, w2h);
  hipLaunchKernelGGL(k_cvt_wo, dim3((OUT_DIM * KTOT) / 1024), dim3(256), 0, stream, wo, woh);
  hipLaunchKernelGGL(k_gen_h, dim3(BT), dim3(256), 0, stream,
                     query, ln_g, ln_b, w1, b1, Hh);

  for (int ch = 0; ch < nch; ++ch) {
    const int tok0 = ch * CHUNK;
    hipLaunchKernelGGL(k_gen_params, dim3(544), dim3(256), 0, stream,
                       w2h, b2, Hh, cmT, smP, tok0);
    hipLaunchKernelGGL(k_mix, dim3(CHUNK), dim3(256), 0, stream,
                       sampled, cmT, smP, m_beta, s_beta, o2t, tok0);
    hipLaunchKernelGGL(k_proj, dim3(32 * 16), dim3(256), 0, stream,
                       o2t, woh, bo, out, tok0);
  }
}
